// LSTMModel_66864050864903
// MI455X (gfx1250) — compile-verified, matrix operand order checked by reading, not run on hardware
//
#include <hip/hip_runtime.h>

// ---------------- problem constants ----------------
#define NWG 8           // workgroups (one WGP each), each owns 32 columns of H
#define BT  512         // threads per block = 16 wave32 waves

constexpr int Bz   = 128;   // batch
constexpr int Tt   = 1024;  // time steps
constexpr int Dd   = 7;     // input features
constexpr int Hh   = 256;   // hidden
constexpr int Zz   = 263;   // H + D
constexpr int ZP   = 288;   // K padded to 9*32 for gate GEMM
constexpr int ZSTR = 304;   // LDS row stride (bf16 elems), rows 32B aligned
constexpr int KO   = 256;   // K for o2 GEMM
constexpr int KOSTR= 272;   // LDS row stride for c-full / W_o2
constexpr int HS   = Hh / NWG;  // 32 columns per WG
constexpr int GN   = 4 * HS;    // 128 gate columns per WG (i1|i2|f|o1)

typedef __attribute__((ext_vector_type(16))) __bf16 v16bf;
typedef __attribute__((ext_vector_type(8)))  float  v8f;
typedef __attribute__((ext_vector_type(4)))  int    v4i;

#define AS1 __attribute__((address_space(1)))
#define AS3 __attribute__((address_space(3)))

// gfx1250 async global->LDS path (ASYNCcnt-tracked), with safe fallback.
#if defined(__has_builtin)
# if __has_builtin(__builtin_amdgcn_global_load_async_to_lds_b128) && \
     __has_builtin(__builtin_amdgcn_s_wait_asynccnt)
#  define HAVE_ASYNC_LDS 1
# endif
#endif
#ifndef HAVE_ASYNC_LDS
# define HAVE_ASYNC_LDS 0
#endif

__device__ __forceinline__ unsigned short f2bf(float f) {
  union { float f; unsigned int u; } v; v.f = f;
  unsigned int u = v.u;
  return (unsigned short)((u + 0x7FFFu + ((u >> 16) & 1u)) >> 16);  // RNE
}
__device__ __forceinline__ float sigm(float x)      { return 1.0f / (1.0f + __expf(-x)); }
__device__ __forceinline__ float tanh_fast(float x) { return 2.0f / (1.0f + __expf(-2.0f * x)) - 1.0f; }

// Device-scope barrier across the 8 persistent blocks (all co-resident).
__device__ __forceinline__ void grid_barrier(unsigned int* bar, unsigned int target) {
  __threadfence();
  __syncthreads();
  if (threadIdx.x == 0) {
    __hip_atomic_fetch_add(bar, 1u, __ATOMIC_RELEASE, __HIP_MEMORY_SCOPE_AGENT);
    while (__hip_atomic_load(bar, __ATOMIC_ACQUIRE, __HIP_MEMORY_SCOPE_AGENT) < target) {
      __builtin_amdgcn_s_sleep(2);
    }
  }
  __syncthreads();
}

__global__ void lstm_bar_init(unsigned int* bar) { bar[threadIdx.x] = 0u; }

__global__ void __launch_bounds__(BT, 1)
lstm_persistent(const float* __restrict__ x,
                const float* __restrict__ Wi1, const float* __restrict__ bi1,
                const float* __restrict__ Wi2, const float* __restrict__ bi2,
                const float* __restrict__ Wf,  const float* __restrict__ bfg,
                const float* __restrict__ Wo1, const float* __restrict__ bo1,
                const float* __restrict__ Wo2, const float* __restrict__ bo2,
                float* __restrict__ out,
                unsigned short* __restrict__ cstage,   // [B][H] bf16 staging in ws
                unsigned short* __restrict__ hstage,   // [B][H] bf16 staging in ws
                unsigned int*   __restrict__ bar)
{
  // -------- LDS (≈269 KB of the WGP's 320 KB) --------
  __shared__ __align__(32) unsigned short zbuf[Bz * ZSTR];   // z=[h|x|0] rows, bf16
  __shared__ __align__(32) unsigned short wgl [GN * ZSTR];   // gate weights^T [n][k], bf16
  __shared__ __align__(32) unsigned short wo2l[HS * KOSTR];  // W_o2^T [n][k], bf16
  __shared__ __align__(32) unsigned char  shbuf[Bz * KOSTR * 2]; // gate preacts f32  ∪  c-full bf16
  __shared__ float cslice[Bz * HS];                          // persistent c (f32) for this WG's cols
  __shared__ float o1buf [Bz * HS];                          // sigmoid(o1) between the two GEMMs

  float*          gatebuf = (float*)shbuf;          // [Bz][GN] f32
  unsigned short* cfull   = (unsigned short*)shbuf; // [Bz][KOSTR] bf16

  const int tid   = threadIdx.x;
  const int wg    = blockIdx.x;
  const int wave  = tid >> 5;
  const int lane  = tid & 31;
  const int l16   = lane & 15;
  const int khalf = (lane >> 4) << 4;   // 0 or 16 (half-wave K split)

  // -------- one-time setup: zero state, stage weights into LDS as bf16 --------
  for (int i = tid; i < Bz * ZSTR; i += BT) zbuf[i] = 0;       // h0 = 0, pads = 0
  for (int i = tid; i < Bz * HS;   i += BT) cslice[i] = 0.0f;  // c0 = 0
  for (int idx = tid; idx < GN * ZP; idx += BT) {
    int n = idx / ZP, k = idx - n * ZP;
    int g = n >> 5, j = n & 31;
    int col = wg * HS + j;
    float v = 0.0f;
    if (k < Zz) {
      const float* W = (g == 0) ? Wi1 : (g == 1) ? Wi2 : (g == 2) ? Wf : Wo1;
      v = W[k * Hh + col];
    }
    wgl[n * ZSTR + k] = f2bf(v);
  }
  for (int idx = tid; idx < HS * KO; idx += BT) {
    int n = idx / KO, k = idx - n * KO;
    wo2l[n * KOSTR + k] = f2bf(Wo2[k * Hh + wg * HS + n]);
  }
  __syncthreads();

  unsigned int bar_target = 0;

  for (int t = 0; t < Tt; ++t) {
    // ---- stream x_t into z columns [H, H+D); prefetch x_{t+1} rows ----
    for (int idx = tid; idx < Bz * Dd; idx += BT) {
      int b = idx / Dd, d = idx - b * Dd;
      zbuf[b * ZSTR + Hh + d] = f2bf(x[((size_t)b * Tt + t) * Dd + d]);
    }
    if (tid < Bz && t + 1 < Tt) {
      __builtin_prefetch(&x[((size_t)tid * Tt + (t + 1)) * Dd], 0, 1);
    }
    __syncthreads();   // also orders last step's h->z writes

    // ---- gate GEMM: [128 x 288] x [288 x 128] over 16 waves ----
    {
      const int mt   = wave & 7;            // M tile 0..7
      const int nb   = (wave >> 3) << 2;    // N tile base 0 or 4 (4 tiles per wave)
      const int arow = (mt * 16 + l16) * ZSTR;
      v8f a0 = {}, a1 = {}, a2 = {}, a3 = {};
      #pragma unroll
      for (int kt = 0; kt < 9; ++kt) {
        const int kk = kt * 32 + khalf;
        v16bf av = *(const v16bf*)&zbuf[arow + kk];
        v16bf b0 = *(const v16bf*)&wgl[((nb + 0) * 16 + l16) * ZSTR + kk];
        v16bf b1 = *(const v16bf*)&wgl[((nb + 1) * 16 + l16) * ZSTR + kk];
        v16bf b2 = *(const v16bf*)&wgl[((nb + 2) * 16 + l16) * ZSTR + kk];
        v16bf b3 = *(const v16bf*)&wgl[((nb + 3) * 16 + l16) * ZSTR + kk];
        a0 = __builtin_amdgcn_wmma_f32_16x16x32_bf16(false, av, false, b0, (short)0, a0, false, false);
        a1 = __builtin_amdgcn_wmma_f32_16x16x32_bf16(false, av, false, b1, (short)0, a1, false, false);
        a2 = __builtin_amdgcn_wmma_f32_16x16x32_bf16(false, av, false, b2, (short)0, a2, false, false);
        a3 = __builtin_amdgcn_wmma_f32_16x16x32_bf16(false, av, false, b3, (short)0, a3, false, false);
      }
      // C layout: lanes 0-15 -> M=r, lanes 16-31 -> M=r+8; N = tile*16 + (lane&15)
      const int rbase = mt * 16 + ((lane >> 4) << 3);
      #pragma unroll
      for (int r = 0; r < 8; ++r) {
        gatebuf[(rbase + r) * GN + (nb + 0) * 16 + l16] = a0[r];
        gatebuf[(rbase + r) * GN + (nb + 1) * 16 + l16] = a1[r];
        gatebuf[(rbase + r) * GN + (nb + 2) * 16 + l16] = a2[r];
        gatebuf[(rbase + r) * GN + (nb + 3) * 16 + l16] = a3[r];
      }
    }
    __syncthreads();

    // ---- pointwise cell update for this WG's 32 columns ----
    for (int idx = tid; idx < Bz * HS; idx += BT) {
      int b = idx / HS, j = idx - b * HS;
      int col = wg * HS + j;
      float i1 = sigm     (gatebuf[b * GN +      j] + bi1[col]);
      float i2 = tanh_fast(gatebuf[b * GN + 32 + j] + bi2[col]);
      float fg = sigm     (gatebuf[b * GN + 64 + j] + bfg[col]);
      float o1 = sigm     (gatebuf[b * GN + 96 + j] + bo1[col]);
      float c  = cslice[idx] * fg + i1 * i2;
      cslice[idx] = c;
      o1buf[idx]  = o1;
      cstage[b * Hh + col] = f2bf(c);            // publish c slice for all WGs
    }
    bar_target += NWG;
    grid_barrier(bar, bar_target);               // all c slices visible; gatebuf reads done

    // ---- pull full c (bf16) into LDS for the o2 GEMM ----
#if HAVE_ASYNC_LDS
    {
      const int nchunk = (Bz * Hh) / 8;          // 4096 x 16B chunks
      for (int c = tid; c < nchunk; c += BT) {
        int b = c >> 5, i = c & 31;              // 32 chunks per row
        const unsigned short* src = cstage + b * Hh + i * 8;
        unsigned short*       dst = cfull + b * KOSTR + i * 8;
        __builtin_amdgcn_global_load_async_to_lds_b128((AS1 v4i*)src, (AS3 v4i*)dst, 0, 0);
      }
      __builtin_amdgcn_s_wait_asynccnt(0);
    }
#else
    for (int i = tid; i < Bz * Hh; i += BT) {
      int b = i >> 8, k = i & 255;
      cfull[b * KOSTR + k] = cstage[i];
    }
#endif
    __syncthreads();

    // ---- o2 GEMM: [128 x 256] x [256 x 32], one 16x16 tile per wave ----
    {
      const int mt   = wave & 7;
      const int nt   = wave >> 3;               // 0..1
      const int arow = (mt * 16 + l16) * KOSTR;
      const int brow = (nt * 16 + l16) * KOSTR;
      v8f acc = {};
      #pragma unroll
      for (int kt = 0; kt < 8; ++kt) {
        const int kk = kt * 32 + khalf;
        v16bf av = *(const v16bf*)&cfull[arow + kk];
        v16bf bv = *(const v16bf*)&wo2l[brow + kk];
        acc = __builtin_amdgcn_wmma_f32_16x16x32_bf16(false, av, false, bv, (short)0, acc, false, false);
      }
      const int rbase = mt * 16 + ((lane >> 4) << 3);
      const int j     = nt * 16 + l16;
      const int col   = wg * HS + j;
      const float b2  = bo2[col];
      #pragma unroll
      for (int r = 0; r < 8; ++r) {
        int row  = rbase + r;
        float o2 = tanh_fast(acc[r] + b2);
        float h  = o1buf[row * HS + j] * o2;
        out[((size_t)row * Tt + t) * Hh + col] = h;   // [B,T,H] output
        hstage[row * Hh + col] = f2bf(h);             // publish h slice
      }
    }
    bar_target += NWG;
    grid_barrier(bar, bar_target);               // all h slices visible

    // ---- h -> z (bf16) for next step; ordered by next iteration's barrier ----
#if HAVE_ASYNC_LDS
    {
      const int nchunk = (Bz * Hh) / 8;
      for (int c = tid; c < nchunk; c += BT) {
        int b = c >> 5, i = c & 31;
        const unsigned short* src = hstage + b * Hh + i * 8;
        unsigned short*       dst = zbuf + b * ZSTR + i * 8;
        __builtin_amdgcn_global_load_async_to_lds_b128((AS1 v4i*)src, (AS3 v4i*)dst, 0, 0);
      }
      __builtin_amdgcn_s_wait_asynccnt(0);
    }
#else
    for (int i = tid; i < Bz * Hh; i += BT) {
      int b = i >> 8, k = i & 255;
      zbuf[b * ZSTR + k] = hstage[i];
    }
#endif
    // next loop iteration's __syncthreads (after x load) orders these writes
  }
}

extern "C" void kernel_launch(void* const* d_in, const int* in_sizes, int n_in,
                              void* d_out, int out_size, void* d_ws, size_t ws_size,
                              hipStream_t stream) {
  (void)in_sizes; (void)n_in; (void)out_size; (void)ws_size;
  const float* x   = (const float*)d_in[0];
  const float* Wi1 = (const float*)d_in[1];
  const float* bi1 = (const float*)d_in[2];
  const float* Wi2 = (const float*)d_in[3];
  const float* bi2 = (const float*)d_in[4];
  const float* Wf  = (const float*)d_in[5];
  const float* bfg = (const float*)d_in[6];
  const float* Wo1 = (const float*)d_in[7];
  const float* bo1 = (const float*)d_in[8];
  const float* Wo2 = (const float*)d_in[9];
  const float* bo2 = (const float*)d_in[10];

  unsigned char*  ws     = (unsigned char*)d_ws;
  unsigned short* cstage = (unsigned short*)ws;             // 65536 B
  unsigned short* hstage = (unsigned short*)(ws + 65536);   // 65536 B
  unsigned int*   bar    = (unsigned int*)(ws + 131072);    // barrier counter

  lstm_bar_init<<<1, 16, 0, stream>>>(bar);
  lstm_persistent<<<NWG, BT, 0, stream>>>(x, Wi1, bi1, Wi2, bi2, Wf, bfg,
                                          Wo1, bo1, Wo2, bo2,
                                          (float*)d_out, cstage, hstage, bar);
}
